// FullModel_18631568130507
// MI455X (gfx1250) — matrix operand order repaired and checked by reading, not compiled, not run on hardware
//
#include <hip/hip_runtime.h>
#include <hip/hip_bf16.h>

// ---------------------------------------------------------------------------
// Fused MLP (GEMM1+ReLU, GEMM2) + 100-iter convex stencil solver + gather.
// MI455X / gfx1250: wave32, WMMA bf16 16x16x32, LDS-staged tiles (224 KB/WG).
// ---------------------------------------------------------------------------

#define B_TOTAL  32768
#define N_INP    256
#define N_HID    1024
#define N_F      256
#define N_ITERS  100
#define BM       64          // batch rows per workgroup
#define NTHREADS 256         // 8 waves of 32

typedef __attribute__((ext_vector_type(16))) __bf16 v16bf;
typedef __attribute__((ext_vector_type(8)))  float  v8f;

union FragU {
    v16bf v;
    uint4 q[2];
};

__device__ __forceinline__ unsigned short f2bf(float f) {
    union { float f; unsigned u; } c;
    c.f = f;
    unsigned u = c.u;
    // round-to-nearest-even f32 -> bf16
    unsigned r = (u + 0x7FFFu + ((u >> 16) & 1u)) >> 16;
    return (unsigned short)r;
}

// A-matrix 16x32 bf16 fragment (ISA 7.12.2 layout):
// lane<16: M=lane, K = {k0..k0+7, k0+16..k0+23}
// lane>=16: M=lane-16, K = {k0+8..k0+15, k0+24..k0+31}
__device__ __forceinline__ v16bf load_a_frag_lds(const unsigned short* S, int ld,
                                                 int m0, int k0, int lane) {
    int r    = m0 + (lane & 15);
    int koff = k0 + ((lane & 16) ? 8 : 0);
    const unsigned short* p = S + r * ld + koff;
    FragU f;
    f.q[0] = *(const uint4*)(p);
    f.q[1] = *(const uint4*)(p + 16);
    return f.v;
}

// B-matrix 32x16 bf16 fragment, built from row-major W[N][K]:
// lane<16: N=lane, K = {k0..k0+7, k0+16..k0+23} ; lane>=16: N=lane-16, K = {k0+8..k0+15, k0+24..k0+31}
__device__ __forceinline__ v16bf load_b_frag(const unsigned short* W, int ld,
                                             int n0, int k0, int lane) {
    int r    = n0 + (lane & 15);
    int koff = k0 + ((lane & 16) ? 8 : 0);
    const unsigned short* p = W + (size_t)r * ld + koff;
    FragU f;
    f.q[0] = *(const uint4*)(p);
    f.q[1] = *(const uint4*)(p + 16);
    return f.v;
}

__device__ __forceinline__ v8f wmma_bf16(v16bf a, v16bf b, v8f c) {
    return __builtin_amdgcn_wmma_f32_16x16x32_bf16(
        /*neg_a=*/false, a, /*neg_b=*/false, b,
        /*c_mod=*/(short)0, c, /*reuse_a=*/false, /*reuse_b=*/false);
}

// --------------------------- weight conversion -----------------------------
__global__ __launch_bounds__(256)
void cvt_weights_kernel(const float* __restrict__ W1, const float* __restrict__ W2,
                        unsigned short* __restrict__ W1b, unsigned short* __restrict__ W2b) {
    int i    = blockIdx.x * blockDim.x + threadIdx.x;  // 131072 threads
    int base = i * 4;
    const int NW1 = N_HID * N_INP;                     // 262144
    if (base < NW1) {
        float4 v = *(const float4*)(W1 + base);
        unsigned short* p = W1b + base;
        p[0] = f2bf(v.x); p[1] = f2bf(v.y); p[2] = f2bf(v.z); p[3] = f2bf(v.w);
    } else {
        int b2 = base - NW1;
        float4 v = *(const float4*)(W2 + b2);
        unsigned short* p = W2b + b2;
        p[0] = f2bf(v.x); p[1] = f2bf(v.y); p[2] = f2bf(v.z); p[3] = f2bf(v.w);
    }
}

// ------------------------------- fused kernel ------------------------------
// LDS layout (dynamic, 229376 bytes):
//   [0,      32768)  sA : bf16 x-tile  [BM][256]
//   [32768, 163840)  sH : bf16 h-tile  [BM][1024]   (reused as stencil bufB)
//   [163840,229376)  sP : f32 param    [BM][256]    (stencil bufA, final result)
extern __shared__ __align__(16) char smem[];

__global__ __launch_bounds__(NTHREADS)
void fused_mlp_convex_kernel(const float* __restrict__ x,
                             const unsigned short* __restrict__ W1b,
                             const float* __restrict__ b1,
                             const unsigned short* __restrict__ W2b,
                             const float* __restrict__ b2,
                             float* __restrict__ out) {
    const int tid  = threadIdx.x;
    const int wid  = tid >> 5;
    const int lane = tid & 31;
    const int g0   = blockIdx.x * BM;

    unsigned short* sA   = (unsigned short*)(smem);
    unsigned short* sH   = (unsigned short*)(smem + 32768);
    float*          sP   = (float*)(smem + 163840);
    float*          bufB = (float*)(smem + 32768);    // aliases sH (dead after GEMM2)

    // ---- Phase 0: load x tile [BM][256], convert f32 -> bf16 into LDS ----
    {
        int r = tid >> 2;                 // 64 rows
        int c = (tid & 3) * 64;           // 4 chunks of 64 features
        const float* xr = x + (size_t)(g0 + r) * (N_INP + 1) + c;
        unsigned short* ar = sA + r * N_INP + c;
        #pragma unroll 8
        for (int i = 0; i < 64; ++i) ar[i] = f2bf(xr[i]);
    }
    __syncthreads();

    // ---- Phase 1: h = relu(x @ W1^T + b1) -> sH (bf16) ----
    // wave 'wid' owns hidden columns [wid*128, wid*128+128)
    for (int nt = 0; nt < 8; ++nt) {
        int n0 = (wid * 8 + nt) * 16;
        v8f acc[4] = {};
        for (int kt = 0; kt < 8; ++kt) {
            v16bf bf = load_b_frag(W1b, N_INP, n0, kt * 32, lane);
            #pragma unroll
            for (int mt = 0; mt < 4; ++mt) {
                v16bf af = load_a_frag_lds(sA, N_INP, mt * 16, kt * 32, lane);
                acc[mt] = wmma_bf16(af, bf, acc[mt]);
            }
        }
        int   ncol = n0 + (lane & 15);
        float bias = b1[ncol];
        int   mb   = (lane & 16) ? 8 : 0;
        #pragma unroll
        for (int mt = 0; mt < 4; ++mt) {
            #pragma unroll
            for (int j = 0; j < 8; ++j) {
                float v = acc[mt][j] + bias;
                v = v > 0.0f ? v : 0.0f;
                sH[(mt * 16 + mb + j) * N_HID + ncol] = f2bf(v);
            }
        }
    }
    __syncthreads();

    // ---- Phase 2: param = h @ W2^T + b2 -> sP (f32) ----
    // wave 'wid' owns output columns [wid*32, wid*32+32)
    for (int nt = 0; nt < 2; ++nt) {
        int n0 = (wid * 2 + nt) * 16;
        v8f acc[4] = {};
        for (int kt = 0; kt < 32; ++kt) {
            v16bf bf = load_b_frag(W2b, N_HID, n0, kt * 32, lane);
            #pragma unroll
            for (int mt = 0; mt < 4; ++mt) {
                v16bf af = load_a_frag_lds(sH, N_HID, mt * 16, kt * 32, lane);
                acc[mt] = wmma_bf16(af, bf, acc[mt]);
            }
        }
        int   ncol = n0 + (lane & 15);
        float bias = b2[ncol];
        int   mb   = (lane & 16) ? 8 : 0;
        #pragma unroll
        for (int mt = 0; mt < 4; ++mt) {
            #pragma unroll
            for (int j = 0; j < 8; ++j) {
                sP[(mt * 16 + mb + j) * N_F + ncol] = acc[mt][j] + bias;
            }
        }
    }
    __syncthreads();

    // ---- Phase 3: convex fixed point, tracked as w = z + param ----
    // w_{t+1}[i] = w[i] - 0.5*max(0, -w[i-1] + 2 w[i] - w[i+1]), i in [1,254];
    // boundaries frozen.  w_0 = param; final output = w_100.
    {
        int r = tid >> 2;
        int c = (tid & 3) * 64;
        float* bufs[2] = { sP, bufB };
        for (int it = 0; it < N_ITERS; ++it) {
            const float* src = bufs[it & 1] + r * N_F;
            float*       dst = bufs[(it + 1) & 1] + r * N_F;
            #pragma unroll 4
            for (int i = 0; i < 64; ++i) {
                int   f  = c + i;
                float wc = src[f];
                float o  = wc;
                if (f > 0 && f < N_F - 1) {
                    float d = 2.0f * wc - src[f - 1] - src[f + 1];
                    if (d > 0.0f) o = wc - 0.5f * d;
                }
                dst[f] = o;
            }
            __syncthreads();
        }
        // N_ITERS even -> final values live in bufs[0] == sP
    }

    // ---- Phase 4: gather at idx = round(query * 255) and store ----
    if (tid < BM) {
        int   g = g0 + tid;
        float q = x[(size_t)g * (N_INP + 1) + N_INP];
        int idx = (int)rintf(q * (float)(N_F - 1));
        idx = idx < 0 ? 0 : (idx > N_F - 1 ? N_F - 1 : idx);
        out[g] = sP[tid * N_F + idx];
    }
}

// ------------------------------- launcher ----------------------------------
extern "C" void kernel_launch(void* const* d_in, const int* in_sizes, int n_in,
                              void* d_out, int out_size, void* d_ws, size_t ws_size,
                              hipStream_t stream) {
    const float* x  = (const float*)d_in[0];   // [32768, 257]
    const float* W1 = (const float*)d_in[1];   // [1024, 256]
    const float* b1 = (const float*)d_in[2];   // [1024]
    const float* W2 = (const float*)d_in[3];   // [256, 1024]
    const float* b2 = (const float*)d_in[4];   // [256]
    float* out = (float*)d_out;                // [32768]

    unsigned short* W1b = (unsigned short*)d_ws;            // 512 KB
    unsigned short* W2b = W1b + (N_HID * N_INP);            // 512 KB

    // Convert weights f32 -> bf16 once per call (524288 elems / 4 per thread).
    cvt_weights_kernel<<<512, 256, 0, stream>>>(W1, W2, W1b, W2b);

    // 512 workgroups x 256 threads; 229376 B dynamic LDS per workgroup.
    const size_t lds_bytes = 229376;
    fused_mlp_convex_kernel<<<B_TOTAL / BM, NTHREADS, lds_bytes, stream>>>(
        x, W1b, b1, W2b, b2, out);
}
